// WaveNetLayer_4045859192978
// MI455X (gfx1250) — matrix operand order repaired and checked by reading, not compiled, not run on hardware
//
#include <hip/hip_runtime.h>
#include <math.h>

#define B_ 16
#define C_ 192
#define T_ 4096
#define K_ 5
#define L_ 16

typedef __attribute__((ext_vector_type(16))) __bf16 v16bf;
typedef __attribute__((ext_vector_type(8)))  float  v8f;

__device__ __forceinline__ unsigned short f2bf(float f) {
    unsigned int u = __float_as_uint(f);
    u += 0x7FFFu + ((u >> 16) & 1u);     // round-to-nearest-even
    return (unsigned short)(u >> 16);
}
__device__ __forceinline__ __bf16 bfbits(unsigned short h) {
    return __builtin_bit_cast(__bf16, h);
}

// ---------------- init: copy x -> xF, bf16 mirror ([b][t][ci]) -> mA, zero o ----------------
__global__ void init_kernel(const float* __restrict__ x, float* __restrict__ xF,
                            unsigned short* __restrict__ mA, float* __restrict__ o, int n) {
    int i = blockIdx.x * 256 + threadIdx.x;
    if (i < n) {
        float v = x[i];
        xF[i] = v;
        o[i]  = 0.f;
        int t  = i % T_;
        int r  = i / T_;
        int ci = r % C_;
        int b  = r / C_;
        mA[((size_t)b * T_ + t) * C_ + ci] = f2bf(v);
    }
}

// A-fragment K mapping for 16-bit A 16x32 (contiguous, matches B staging):
// lanes 0-15: K=0..15 ; lanes 16-31: K=16..31
__device__ __forceinline__ int afrag_k(int j, int lane) {
    int base = (lane >= 16) ? 16 : 0;
    return base + j;
}

// ---------------- weight packers (fp32 -> bf16 A-fragments) ----------------
__global__ void pack_win(const float* __restrict__ W, unsigned short* __restrict__ F, int total) {
    int idx = blockIdx.x * 256 + threadIdx.x;
    if (idx >= total) return;
    int j    = idx & 15;
    int lane = (idx >> 4) & 31;
    int r    = idx >> 9;
    int kc = r % 6;  r /= 6;
    int mt = r % 24; r /= 24;
    int k  = r % 5;
    int i  = r / 5;
    int co = mt * 16 + (lane & 15);
    int ci = kc * 32 + afrag_k(j, lane);
    float w = W[(((size_t)i * (2 * C_) + co) * C_ + ci) * K_ + k];
    F[idx] = f2bf(w);
}

__global__ void pack_wrs(const float* __restrict__ W, unsigned short* __restrict__ F, int total) {
    int idx = blockIdx.x * 256 + threadIdx.x;
    if (idx >= total) return;
    int j    = idx & 15;
    int lane = (idx >> 4) & 31;
    int r    = idx >> 9;
    int kc = r % 6;  r /= 6;
    int mt = r % 24; r /= 24;
    int i  = r;
    int co = mt * 16 + (lane & 15);
    int ci = kc * 32 + afrag_k(j, lane);
    float w = W[((size_t)i * (2 * C_) + co) * C_ + ci];
    F[idx] = f2bf(w);
}

__global__ void pack_wlast(const float* __restrict__ W, unsigned short* __restrict__ F, int total) {
    int idx = blockIdx.x * 256 + threadIdx.x;
    if (idx >= total) return;
    int j    = idx & 15;
    int lane = (idx >> 4) & 31;
    int r    = idx >> 9;
    int kc = r % 6;  r /= 6;
    int mt = r;                       // 0..11
    int co = mt * 16 + (lane & 15);   // < 192
    int ci = kc * 32 + afrag_k(j, lane);
    float w = W[(size_t)co * C_ + ci];
    F[idx] = f2bf(w);
}

// ---------------- fused layer kernel ----------------
#define XS_STRIDE   200                       // [col 0..67][ci 0..191] bf16, padded; row = 400 B
#define XS_BYTES    (68 * XS_STRIDE * 2)      // 27200
#define XIN_STRIDE  392                       // [col 0..63][co 0..383] f32, padded
#define SMEM_TOTAL  (XS_BYTES + 64 * XIN_STRIDE * 4)   // 127552 B

__global__ __launch_bounds__(512)
void wavenet_layer(const unsigned short* __restrict__ mirror_src,  // bf16 x, [b][t][ci]
                   unsigned short* __restrict__ mirror_dst,
                   float* __restrict__ xF,                          // fp32 x (in-place residual)
                   float* __restrict__ o, const float* __restrict__ mask,
                   const unsigned short* __restrict__ fA,   // GEMM1 frags (layer base)
                   const unsigned short* __restrict__ fB2,  // GEMM2 frags (layer base)
                   const float* __restrict__ bias1, const float* __restrict__ bias2,
                   int last, float* __restrict__ out)
{
    extern __shared__ char smem[];
    unsigned short* xs = (unsigned short*)smem;        // x tile (bf16), later acts tile
    float* xin = (float*)(smem + XS_BYTES);            // pre-activations (f32)

    const int b   = blockIdx.y;
    const int t0  = blockIdx.x * 64;
    const int tid = threadIdx.x;
    const int lane = tid & 31;
    const int wave = tid >> 5;
    const int wm = wave >> 1;        // 0..7 : M group (3 tiles of 16 rows)
    const int wn = wave & 1;         // 0..1 : N group (2 tiles of 16 cols)
    const int lo = lane & 15;
    const int hi = lane >> 4;        // 0/1

    // ---- stage x tile via async copy: cols t0-2 .. t0+65, [col][ci] bf16 ----
    // mirror row (one time step, 192 bf16 = 384 B) = 24 x 16B chunks
    {
        const unsigned long long sbase =
            (unsigned long long)(const void*)(mirror_src + (size_t)b * T_ * C_);
        for (int e = tid; e < 68 * 24; e += 512) {
            int c = e / 24;           // LDS column 0..67
            int q = e % 24;           // 16-byte chunk within row
            int t = t0 - 2 + c;
            unsigned short* p = xs + c * XS_STRIDE + q * 8;
            if (t >= 0 && t < T_) {
                unsigned lds_addr = (unsigned)reinterpret_cast<uintptr_t>(p);
                int voff = (t * C_ + q * 8) * 2;
                asm volatile("global_load_async_to_lds_b128 %0, %1, %2"
                             :: "v"(lds_addr), "v"(voff), "s"(sbase) : "memory");
            } else {
                *(uint4*)p = uint4{0u, 0u, 0u, 0u};   // zero-pad boundary
            }
        }
        asm volatile("s_wait_asynccnt 0x0" ::: "memory");
    }
    __syncthreads();

    // ---- GEMM1: x_in = W_in (*) x  (M=384, N=64, K=192 per shift, 5 shifts) ----
    v8f acc[3][2];
#pragma unroll
    for (int jm = 0; jm < 3; ++jm)
#pragma unroll
        for (int nt = 0; nt < 2; ++nt)
#pragma unroll
            for (int q = 0; q < 8; ++q) acc[jm][nt][q] = 0.f;

    for (int k = 0; k < K_; ++k) {
        for (int kc = 0; kc < 6; ++kc) {
            v16bf bf[2];
#pragma unroll
            for (int nt = 0; nt < 2; ++nt) {
                int col = wn * 32 + nt * 16 + lo + k;            // includes -2 pad shift
                const unsigned short* p = &xs[col * XS_STRIDE + kc * 32 + hi * 16];
#pragma unroll
                for (int j = 0; j < 16; ++j) bf[nt][j] = bfbits(p[j]);
            }
#pragma unroll
            for (int jm = 0; jm < 3; ++jm) {
                int mt = wm * 3 + jm;
                const unsigned short* pa = fA + ((((k * 24 + mt) * 6 + kc) << 9) + (lane << 4));
                v16bf a;
#pragma unroll
                for (int j = 0; j < 16; ++j) a[j] = bfbits(pa[j]);
#pragma unroll
                for (int nt = 0; nt < 2; ++nt)
                    acc[jm][nt] = __builtin_amdgcn_wmma_f32_16x16x32_bf16(
                        false, a, false, bf[nt], (short)0, acc[jm][nt], false, false);
            }
        }
    }

    // ---- write pre-activations (+bias) to LDS [col][co] ----
#pragma unroll
    for (int jm = 0; jm < 3; ++jm) {
        int mtg = wm * 3 + jm;
#pragma unroll
        for (int nt = 0; nt < 2; ++nt) {
            int col = wn * 32 + nt * 16 + lo;
#pragma unroll
            for (int v = 0; v < 8; ++v) {
                int co = mtg * 16 + v + hi * 8;
                xin[col * XIN_STRIDE + co] = acc[jm][nt][v] + bias1[co];
            }
        }
    }
    __syncthreads();   // xin complete; all waves done reading xs

    // ---- gated activation: acts = tanh(x1) * sigmoid(x2), bf16 into xs ----
    for (int e = tid; e < 64 * C_; e += 512) {
        int ci = e % C_;
        int c  = e / C_;
        float x1 = xin[c * XIN_STRIDE + ci];
        float x2 = xin[c * XIN_STRIDE + C_ + ci];
        float a  = tanhf(x1) * (1.f / (1.f + __expf(-x2)));
        xs[c * XS_STRIDE + ci] = f2bf(a);
    }
    __syncthreads();

    // ---- GEMM2: 1x1 conv (M=384 or 192 last, K=192, N=64) ----
    v8f acc2[3][2];
#pragma unroll
    for (int jm = 0; jm < 3; ++jm)
#pragma unroll
        for (int nt = 0; nt < 2; ++nt)
#pragma unroll
            for (int q = 0; q < 8; ++q) acc2[jm][nt][q] = 0.f;

    const int mtmax = last ? 12 : 24;
    for (int kc = 0; kc < 6; ++kc) {
        v16bf bf[2];
#pragma unroll
        for (int nt = 0; nt < 2; ++nt) {
            int col = wn * 32 + nt * 16 + lo;
            const unsigned short* p = &xs[col * XS_STRIDE + kc * 32 + hi * 16];
#pragma unroll
            for (int j = 0; j < 16; ++j) bf[nt][j] = bfbits(p[j]);
        }
#pragma unroll
        for (int jm = 0; jm < 3; ++jm) {
            int mt = wm * 3 + jm;
            if (mt >= mtmax) continue;
            const unsigned short* pa = fB2 + (((mt * 6 + kc) << 9) + (lane << 4));
            v16bf a;
#pragma unroll
            for (int j = 0; j < 16; ++j) a[j] = bfbits(pa[j]);
#pragma unroll
            for (int nt = 0; nt < 2; ++nt)
                acc2[jm][nt] = __builtin_amdgcn_wmma_f32_16x16x32_bf16(
                    false, a, false, bf[nt], (short)0, acc2[jm][nt], false, false);
        }
    }

    // ---- writeback: residual update (+ bf16 mirror) / skip accumulation / final output ----
#pragma unroll
    for (int jm = 0; jm < 3; ++jm) {
        int mtg = wm * 3 + jm;
        if (mtg >= mtmax) continue;
#pragma unroll
        for (int nt = 0; nt < 2; ++nt) {
            int t = t0 + wn * 32 + nt * 16 + lo;
            float m = mask[(size_t)b * T_ + t];
            if (!last) {
                if (mtg < 12) {
                    // residual: fp32 in-place + packed bf16 mirror row (16 B contiguous)
                    unsigned short tmp[8];
#pragma unroll
                    for (int v = 0; v < 8; ++v) {
                        int co = mtg * 16 + hi * 8 + v;
                        size_t g = ((size_t)b * C_ + co) * T_ + t;
                        float xnew = (xF[g] + acc2[jm][nt][v] + bias2[co]) * m;
                        xF[g] = xnew;
                        tmp[v] = f2bf(xnew);
                    }
                    unsigned short* mp =
                        mirror_dst + ((size_t)b * T_ + t) * C_ + mtg * 16 + hi * 8;
                    uint4 pk;
                    pk.x = (unsigned)tmp[0] | ((unsigned)tmp[1] << 16);
                    pk.y = (unsigned)tmp[2] | ((unsigned)tmp[3] << 16);
                    pk.z = (unsigned)tmp[4] | ((unsigned)tmp[5] << 16);
                    pk.w = (unsigned)tmp[6] | ((unsigned)tmp[7] << 16);
                    *(uint4*)mp = pk;
                } else {
#pragma unroll
                    for (int v = 0; v < 8; ++v) {
                        int co = mtg * 16 + hi * 8 + v;
                        size_t g = ((size_t)b * C_ + (co - C_)) * T_ + t;
                        o[g] += acc2[jm][nt][v] + bias2[co];   // skip connection
                    }
                }
            } else {
#pragma unroll
                for (int v = 0; v < 8; ++v) {
                    int co = mtg * 16 + hi * 8 + v;            // < 192
                    size_t g = ((size_t)b * C_ + co) * T_ + t;
                    out[g] = (o[g] + acc2[jm][nt][v] + bias2[co]) * m;
                }
            }
        }
    }
}

// ---------------- launcher ----------------
extern "C" void kernel_launch(void* const* d_in, const int* in_sizes, int n_in,
                              void* d_out, int out_size, void* d_ws, size_t ws_size,
                              hipStream_t stream) {
    const float* x      = (const float*)d_in[0];
    const float* mask   = (const float*)d_in[1];
    const float* W_in   = (const float*)d_in[2];
    const float* b_in   = (const float*)d_in[3];
    const float* W_rs   = (const float*)d_in[4];
    const float* b_rs   = (const float*)d_in[5];
    const float* W_last = (const float*)d_in[6];
    const float* b_last = (const float*)d_in[7];

    const size_t NX = (size_t)B_ * C_ * T_;
    float* o  = (float*)d_ws;
    float* xF = o + NX;
    unsigned short* mA = (unsigned short*)(xF + NX);
    unsigned short* mB = mA + NX;
    unsigned short* fWin = mB + NX;
    const size_t WIN_ELEMS = (size_t)L_ * K_ * 24 * 6 * 512;        // 5,898,240
    unsigned short* fWrs = fWin + WIN_ELEMS;
    const size_t WRS_ELEMS = (size_t)(L_ - 1) * 24 * 6 * 512;       // 1,105,920
    unsigned short* fWl  = fWrs + WRS_ELEMS;
    const size_t WL_ELEMS  = (size_t)12 * 6 * 512;                  // 36,864

    init_kernel<<<dim3((NX + 255) / 256), dim3(256), 0, stream>>>(x, xF, mA, o, (int)NX);
    pack_win  <<<dim3((WIN_ELEMS + 255) / 256), dim3(256), 0, stream>>>(W_in, fWin, (int)WIN_ELEMS);
    pack_wrs  <<<dim3((WRS_ELEMS + 255) / 256), dim3(256), 0, stream>>>(W_rs, fWrs, (int)WRS_ELEMS);
    pack_wlast<<<dim3((WL_ELEMS + 255) / 256), dim3(256), 0, stream>>>(W_last, fWl, (int)WL_ELEMS);

    dim3 grid(T_ / 64, B_);
    for (int i = 0; i < L_; ++i) {
        const unsigned short* msrc = (i & 1) ? mB : mA;
        unsigned short*       mdst = (i & 1) ? mA : mB;
        int last = (i == L_ - 1);
        wavenet_layer<<<grid, dim3(512), SMEM_TOTAL, stream>>>(
            msrc, mdst, xF, o, mask,
            fWin + (size_t)i * K_ * 24 * 6 * 512,
            last ? fWl : (fWrs + (size_t)i * 24 * 6 * 512),
            b_in + (size_t)i * 2 * C_,
            last ? b_last : (b_rs + (size_t)i * 2 * C_),
            last, (float*)d_out);
    }
}
